// MultiheadAttention_9010841387593
// MI455X (gfx1250) — matrix operand order repaired and checked by reading, not compiled, not run on hardware
//
#include <hip/hip_runtime.h>
#include <hip/hip_bf16.h>

// ---------------------------------------------------------------------------
// MI455X (gfx1250) multi-head attention forward.
// B=4, S=2048, D=1024, H=16, DEPTH=64.
// All GEMMs run through v_wmma_f32_16x16x32_bf16 (wave32).
// ---------------------------------------------------------------------------

#define Bsz   4
#define Ssz   2048
#define Dsz   1024
#define Hsz   16
#define DEPTH 64

typedef __attribute__((ext_vector_type(16))) __bf16 v16bf;
typedef __attribute__((ext_vector_type(8)))  __bf16 v8bf;
typedef __attribute__((ext_vector_type(8)))  float  v8f;
typedef __attribute__((ext_vector_type(4)))  float  v4f;

#if defined(__has_builtin)
#if __has_builtin(__builtin_amdgcn_global_load_async_to_lds_b128)
#define HAVE_ASYNC_LDS 1
#endif
#endif

#ifdef HAVE_ASYNC_LDS
// Builtin expects pointers to 128-bit int vectors: (global v4i*, LDS v4i*, Ii, Ii)
typedef int v4i_t __attribute__((vector_size(16)));
typedef v4i_t __attribute__((address_space(1)))* gptr4_t;
typedef v4i_t __attribute__((address_space(3)))* lptr4_t;
#endif

// --- WMMA wrapper ----------------------------------------------------------
__device__ __forceinline__ v8f wmma_bf16(v16bf a, v16bf b, v8f c) {
  // (neg_a, A, neg_b, B, c_mod, C, reuse_a, reuse_b)
  return __builtin_amdgcn_wmma_f32_16x16x32_bf16(false, a, false, b,
                                                 (short)0, c, false, false);
}

// --- A fragment (16x32 bf16): elements 0..7  -> K = kk + half*8 + e
//                              elements 8..15 -> K = kk + 16 + half*8 + e
__device__ __forceinline__ v16bf load_a_frag(const __bf16* row, int kk, int half) {
  union { v16bf v; v8bf h[2]; } u;
  u.h[0] = *(const v8bf*)(row + kk + half * 8);
  u.h[1] = *(const v8bf*)(row + kk + 16 + half * 8);
  return u.v;
}

// --- B fragment (32x16 bf16): lane holds column N; elements 0..7 -> K = kk + half*8 + e,
//                              elements 8..15 -> K = kk + 16 + half*8 + e
__device__ __forceinline__ v16bf load_b_frag(const __bf16* col, int kk, int half) {
  union { v16bf v; v8bf h[2]; } u;
  u.h[0] = *(const v8bf*)(col + kk + half * 8);
  u.h[1] = *(const v8bf*)(col + kk + 16 + half * 8);
  return u.v;
}

// --- A fragment built from an fp32 row (LDS softmax weights) ---------------
__device__ __forceinline__ v16bf a_frag_from_f32(const float* row, int kk, int half) {
  const float* c0 = row + kk + half * 8;
  const float* c1 = row + kk + 16 + half * 8;
  v16bf r;
#pragma unroll
  for (int e = 0; e < 8; ++e) {
    r[e]     = (__bf16)c0[e];
    r[8 + e] = (__bf16)c1[e];
  }
  return r;
}

// ===========================================================================
// Kernel 1: projection.  X (f32, [B*S,D]) @ W + bias -> bf16 head-split.
// VMODE==0 : O[b][h][s][64]   (q, k)
// VMODE==1 : O[b][h][64][s]   (v, transposed for phase-2 B fragments)
// grid (D/64, B*S/64), 256 thr = 8 waves; each wave: 16x32 tile (2 WMMA/step)
// ===========================================================================
template <int VMODE>
__global__ __launch_bounds__(256)
void proj_qkv_kernel(const float* __restrict__ X, const float* __restrict__ W,
                     const float* __restrict__ bias, __bf16* __restrict__ O) {
  __shared__ __bf16 As[64 * 32];    // 64 M x 32 K
  __shared__ __bf16 WsT[64 * 32];   // 64 N x 32 K (transposed W tile)

  const int t = threadIdx.x;
  const int lane = t & 31, wid = t >> 5;
  const int r = lane & 15, half = lane >> 4;
  const int m0 = blockIdx.y * 64;
  const int n0 = blockIdx.x * 64;
  const int wm = wid & 3, wn = wid >> 2;

  const int arow = t >> 2, aseg = (t & 3) * 8;   // A stage: 4 thr/row, 8 f32
  const int wrow = t >> 3, wseg = (t & 7) * 8;   // W stage: 8 thr/row, 8 f32

  v8f acc0 = {}, acc1 = {};
  for (int kk = 0; kk < Dsz; kk += 32) {
    const float* ap = X + (size_t)(m0 + arow) * Dsz + kk + aseg;
    __builtin_prefetch(ap + 32, 0, 0);           // speculative, safe past end
#pragma unroll
    for (int j = 0; j < 8; ++j) As[arow * 32 + aseg + j] = (__bf16)ap[j];
    const float* wp = W + (size_t)(kk + wrow) * Dsz + n0 + wseg;
#pragma unroll
    for (int j = 0; j < 8; ++j) WsT[(wseg + j) * 32 + wrow] = (__bf16)wp[j];
    __syncthreads();

    const v16bf af  = load_a_frag(As + (wm * 16 + r) * 32, 0, half);
    const v16bf bf0 = load_b_frag(WsT + (wn * 32 + r) * 32, 0, half);
    const v16bf bf1 = load_b_frag(WsT + (wn * 32 + 16 + r) * 32, 0, half);
    acc0 = wmma_bf16(af, bf0, acc0);
    acc1 = wmma_bf16(af, bf1, acc1);
    __syncthreads();
  }

#pragma unroll
  for (int sub = 0; sub < 2; ++sub) {
    const v8f acc = sub ? acc1 : acc0;
    const int gn = n0 + wn * 32 + sub * 16 + r;
    const float bvv = bias[gn];
    const int h = gn >> 6, dep = gn & 63;
#pragma unroll
    for (int i = 0; i < 8; ++i) {
      const int gm = m0 + wm * 16 + i + 8 * half;
      const int b = gm >> 11, s = gm & (Ssz - 1);
      const __bf16 val = (__bf16)(acc[i] + bvv);
      if (VMODE)
        O[(((size_t)(b * Hsz + h)) * DEPTH + dep) * Ssz + s] = val;
      else
        O[(((size_t)(b * Hsz + h)) * Ssz + s) * DEPTH + dep] = val;
    }
  }
}

// ===========================================================================
// Kernel 2: attention for one (b, h, 16-row q tile).
// grid (H, S/16, B): h fastest so 16 WGs sharing a mask strip hit L2.
// ===========================================================================
__global__ __launch_bounds__(256)
void attn_kernel(const __bf16* __restrict__ qb, const __bf16* __restrict__ kb,
                 const __bf16* __restrict__ vt, const float* __restrict__ mask,
                 float* __restrict__ attnw, __bf16* __restrict__ cat) {
  extern __shared__ float smem[];
  float* logits  = smem;                 // 16 * 2048
  float* aux     = logits + 16 * Ssz;    // 256
  float* rowmax  = aux + 256;            // 16
  float* rowinv  = rowmax + 16;          // 16
  float* rowok   = rowinv + 16;          // 16 (+16 pad)
  float* partial = rowok + 32;           // 4 * 256

  const int h = blockIdx.x, qt = blockIdx.y, b = blockIdx.z;
  const int t = threadIdx.x, lane = t & 31, wid = t >> 5;
  const int r = lane & 15, half = lane >> 4;
  const size_t bh = (size_t)(b * Hsz + h);

  // q fragments for this 16-row tile (cached in registers, reused 16x)
  const __bf16* qrow = qb + (bh * Ssz + qt * 16 + r) * DEPTH;
  const v16bf aq0 = load_a_frag(qrow, 0, half);
  const v16bf aq1 = load_a_frag(qrow, 32, half);

  // ---- phase 1: logits strip ----
  for (int kt = 0; kt < 16; ++kt) {
    const int key0 = wid * 256 + kt * 16;
    const __bf16* krow = kb + (bh * Ssz + key0 + r) * DEPTH;
    const v16bf bk0 = load_b_frag(krow, 0, half);
    const v16bf bk1 = load_b_frag(krow, 32, half);
    v8f acc = {};
    acc = wmma_bf16(aq0, bk0, acc);
    acc = wmma_bf16(aq1, bk1, acc);
#pragma unroll
    for (int i = 0; i < 8; ++i) {
      const int qr = i + 8 * half;
      const float m = mask[((size_t)b * Ssz + qt * 16 + qr) * Ssz + key0 + r];
      logits[qr * Ssz + key0 + r] = acc[i] * 0.125f + (1.0f - m) * (-1.0e10f);
    }
  }
  __syncthreads();

  // ---- softmax over 2048 keys, 16 threads per row, float4-vectorized ----
  const int row = t >> 4, sub = t & 15;
  float* lrow = logits + row * Ssz;
  v4f* lrow4 = (v4f*)lrow;

  v4f mx4 = lrow4[sub];
  for (int c = sub + 16; c < Ssz / 4; c += 16) {
    const v4f x = lrow4[c];
#pragma unroll
    for (int j = 0; j < 4; ++j) mx4[j] = fmaxf(mx4[j], x[j]);
  }
  aux[row * 16 + sub] = fmaxf(fmaxf(mx4[0], mx4[1]), fmaxf(mx4[2], mx4[3]));
  __syncthreads();
  if (sub == 0) {
    float m2 = aux[row * 16];
    for (int j = 1; j < 16; ++j) m2 = fmaxf(m2, aux[row * 16 + j]);
    rowmax[row] = m2;
  }
  __syncthreads();
  const float rm = rowmax[row];
  v4f s4 = {};
  for (int c = sub; c < Ssz / 4; c += 16) {
    const v4f x = lrow4[c];
    v4f e;
#pragma unroll
    for (int j = 0; j < 4; ++j) e[j] = __expf(x[j] - rm);
    lrow4[c] = e;
    s4 += e;
  }
  aux[row * 16 + sub] = (s4[0] + s4[1]) + (s4[2] + s4[3]);
  __syncthreads();
  if (sub == 0) {
    float s2 = 0.f;
    for (int j = 0; j < 16; ++j) s2 += aux[row * 16 + j];
    const float ok = (rm > -1.0e9f) ? 1.0f : 0.0f;  // another_mask
    rowok[row]  = ok;
    rowinv[row] = ok / fmaxf(s2, 1e-30f);
  }
  __syncthreads();
  const float inv = rowinv[row];
  v4f* wout4 = (v4f*)(attnw + (bh * Ssz + qt * 16 + row) * Ssz);
  for (int c = sub; c < Ssz / 4; c += 16) {
    const v4f wv = lrow4[c] * inv;
    lrow4[c] = wv;                               // keep weights in LDS
    __builtin_nontemporal_store(wv, wout4 + c);  // 1 GB stream: b128 NT stores
  }
  __syncthreads();

  // ---- phase 2: weights @ v (split-K across wave pairs) ----
  const int nt = wid & 3, khalf = wid >> 2;
  const int dep0 = nt * 16;
  v8f acc = {};
  const float* wrow = logits + r * Ssz;
  const __bf16* vrow = vt + (bh * DEPTH + dep0 + r) * Ssz;  // transposed V
  for (int s = 0; s < 32; ++s) {
    const int kk = khalf * 1024 + s * 32;
    const v16bf afr = a_frag_from_f32(wrow, kk, half);
    const v16bf bfr = load_b_frag(vrow, kk, half);          // two 16B loads
    acc = wmma_bf16(afr, bfr, acc);
  }
  if (khalf == 1) {
#pragma unroll
    for (int i = 0; i < 8; ++i) partial[nt * 256 + lane * 8 + i] = acc[i];
  }
  __syncthreads();
  if (khalf == 0) {
#pragma unroll
    for (int i = 0; i < 8; ++i) {
      const int qr = i + 8 * half;
      const float val = (acc[i] + partial[nt * 256 + lane * 8 + i]) * rowok[qr];
      cat[((size_t)b * Ssz + qt * 16 + qr) * Dsz + h * DEPTH + dep0 + r] = (__bf16)val;
    }
  }
}

// ===========================================================================
// Kernel 3: output = concat (bf16, [B*S,D]) @ Wo (f32) + bo -> f32 d_out.
// A tile staged to LDS with gfx1250 async-to-LDS DMA when available.
// ===========================================================================
__global__ __launch_bounds__(256)
void proj_out_kernel(const __bf16* __restrict__ A, const float* __restrict__ W,
                     const float* __restrict__ bias, float* __restrict__ O) {
  __shared__ __bf16 As[64 * 32];
  __shared__ __bf16 WsT[64 * 32];

  const int t = threadIdx.x;
  const int lane = t & 31, wid = t >> 5;
  const int r = lane & 15, half = lane >> 4;
  const int m0 = blockIdx.y * 64;
  const int n0 = blockIdx.x * 64;
  const int wm = wid & 3, wn = wid >> 2;

  const int arow = t >> 2, aseg = (t & 3) * 8;   // one 16B chunk per thread
  const int wrow = t >> 3, wseg = (t & 7) * 8;

  v8f acc0 = {}, acc1 = {};
  for (int kk = 0; kk < Dsz; kk += 32) {
    const __bf16* ap = A + (size_t)(m0 + arow) * Dsz + kk + aseg;
    __builtin_prefetch(ap + 32, 0, 0);
#ifdef HAVE_ASYNC_LDS
    __builtin_amdgcn_global_load_async_to_lds_b128(
        (gptr4_t)ap, (lptr4_t)(As + arow * 32 + aseg), 0, 0);
#else
    *(uint4*)(As + arow * 32 + aseg) = *(const uint4*)ap;
#endif
    const float* wp = W + (size_t)(kk + wrow) * Dsz + n0 + wseg;
#pragma unroll
    for (int j = 0; j < 8; ++j) WsT[(wseg + j) * 32 + wrow] = (__bf16)wp[j];
#ifdef HAVE_ASYNC_LDS
#if __has_builtin(__builtin_amdgcn_s_wait_asynccnt)
    __builtin_amdgcn_s_wait_asynccnt(0);
#else
    asm volatile("s_wait_asynccnt 0" ::: "memory");
#endif
#endif
    __syncthreads();

    const v16bf af  = load_a_frag(As + (wm * 16 + r) * 32, 0, half);
    const v16bf bf0 = load_b_frag(WsT + (wn * 32 + r) * 32, 0, half);
    const v16bf bf1 = load_b_frag(WsT + (wn * 32 + 16 + r) * 32, 0, half);
    acc0 = wmma_bf16(af, bf0, acc0);
    acc1 = wmma_bf16(af, bf1, acc1);
    __syncthreads();
  }

#pragma unroll
  for (int sub = 0; sub < 2; ++sub) {
    const v8f acc = sub ? acc1 : acc0;
    const int gn = n0 + wn * 32 + sub * 16 + r;
    const float bvv = bias[gn];
#pragma unroll
    for (int i = 0; i < 8; ++i) {
      const int gm = m0 + wm * 16 + i + 8 * half;
      O[(size_t)gm * Dsz + gn] = acc[i] + bvv;
    }
  }
}

// ===========================================================================
extern "C" void kernel_launch(void* const* d_in, const int* in_sizes, int n_in,
                              void* d_out, int out_size, void* d_ws, size_t ws_size,
                              hipStream_t stream) {
  const float* value = (const float*)d_in[0];
  const float* key   = (const float*)d_in[1];
  const float* query = (const float*)d_in[2];
  const float* mask  = (const float*)d_in[3];
  const float* Wq = (const float*)d_in[4];
  const float* bq = (const float*)d_in[5];
  const float* Wk = (const float*)d_in[6];
  const float* bk = (const float*)d_in[7];
  const float* Wv = (const float*)d_in[8];
  const float* bv = (const float*)d_in[9];
  const float* Wo = (const float*)d_in[10];
  const float* bo = (const float*)d_in[11];

  // workspace: q/k head-split bf16, v head-split TRANSPOSED bf16, bf16 concat
  const size_t perBuf = (size_t)Bsz * Hsz * Ssz * DEPTH;  // 8,388,608 elems
  __bf16* qb  = (__bf16*)d_ws;
  __bf16* kb  = qb + perBuf;
  __bf16* vt  = kb + perBuf;
  __bf16* cat = vt + perBuf;

  float* out   = (float*)d_out;                      // (B,S,D)
  float* attnw = out + (size_t)Bsz * Ssz * Dsz;      // (B,H,S,S)

  const dim3 blk(256);
  const dim3 gproj(Dsz / 64, (Bsz * Ssz) / 64);      // (16, 128)

  proj_qkv_kernel<0><<<gproj, blk, 0, stream>>>(query, Wq, bq, qb);
  proj_qkv_kernel<0><<<gproj, blk, 0, stream>>>(key,   Wk, bk, kb);
  proj_qkv_kernel<1><<<gproj, blk, 0, stream>>>(value, Wv, bv, vt);

  // dynamic LDS: 16*2048 logits + 256 aux + 64 rowstats + 1024 partial (f32)
  const size_t smemBytes = (size_t)(16 * Ssz + 256 + 64 + 1024) * sizeof(float);
  attn_kernel<<<dim3(Hsz, Ssz / 16, Bsz), blk, smemBytes, stream>>>(
      qb, kb, vt, mask, attnw, cat);

  proj_out_kernel<<<gproj, blk, 0, stream>>>(cat, Wo, bo, out);
}
